// MultiHeadAttention_29618094474176
// MI455X (gfx1250) — matrix operand order repaired and checked by reading, not compiled, not run on hardware
//
#include <hip/hip_runtime.h>
#include <hip/hip_bf16.h>

// ---------- types ----------
typedef __bf16 v16bf __attribute__((ext_vector_type(16)));
typedef __bf16 bfx8  __attribute__((ext_vector_type(8)));
typedef float  v8f   __attribute__((ext_vector_type(8)));

#define DEVI static __device__ __forceinline__

DEVI bfx8 ldb8(const __bf16* p) { return *(const bfx8*)p; }
DEVI void stb8(__bf16* p, bfx8 v) { *(bfx8*)p = v; }

DEVI v16bf pack16(bfx8 lo, bfx8 hi) {
  v16bf r;
#pragma unroll
  for (int i = 0; i < 8; ++i) { r[i] = lo[i]; r[i + 8] = hi[i]; }
  return r;
}

DEVI v8f wmma_bf16(v16bf a, v16bf b, v8f c) {
  // (neg_a, A, neg_b, B, c_mod, C, reuse_a, reuse_b)
  return __builtin_amdgcn_wmma_f32_16x16x32_bf16(false, a, false, b, (short)0, c,
                                                 false, false);
}

// Async copy 16B from global to LDS (per lane). gfx1250 GLOBAL_LOAD_ASYNC_TO_LDS_B128.
DEVI void async_g2l_b128(const void* gptr, const void* ldsptr) {
  unsigned lds32 = (unsigned)(size_t)ldsptr;             // AS3 offset = low 32 bits
  unsigned long long g64 = (unsigned long long)(size_t)gptr;
  asm volatile("global_load_async_to_lds_b128 %0, %1, off"
               :: "v"(lds32), "v"(g64) : "memory");
}
DEVI void wait_async0() { asm volatile("s_wait_asynccnt 0x0" ::: "memory"); }
DEVI void wait_ds0()    { asm volatile("s_wait_dscnt 0x0" ::: "memory"); }

// ---------- constants ----------
constexpr int Bc = 2, Sc = 2048, Dc = 1024, Hc = 16, HDc = 64;

// ---------- cast fp32 -> bf16 ----------
__global__ void cast_f32_to_bf16(const float* __restrict__ in,
                                 __bf16* __restrict__ out, int n) {
  int i = blockIdx.x * blockDim.x + threadIdx.x;
  if (i < n) out[i] = (__bf16)in[i];
}

// ---------- GEMM: out[M,N] = A[M,K] * W[N,K]^T + bias[N] ----------
// M=4096, N=1024, K=1024. Block = 256 threads = 8 waves.
// Block tile: 256(M) x 64(N). Wave tile: 32(M) x 64(N).
// W tile (64x32) double-buffered in LDS via async-to-LDS copies.
__global__ __launch_bounds__(256) void gemm_nt_bf16(
    const __bf16* __restrict__ A, const __bf16* __restrict__ W,
    const float* __restrict__ bias, float* __restrict__ outF,
    __bf16* __restrict__ outB, int bf16out) {
  constexpr int Kd = 1024, Nd = 1024;
  __shared__ __bf16 ldsW[2][64][40];  // double buffer: 64 N rows x 32 K (+pad)

  const int tid = threadIdx.x;
  const int w = tid >> 5;
  const int lane = tid & 31;
  const int ln = lane & 15;
  const int hi = lane >> 4;
  const int nBase = blockIdx.x * 64;
  const int mRow0 = blockIdx.y * 256 + w * 32;

  v8f zero = {};
  v8f acc[2][4];
#pragma unroll
  for (int hf = 0; hf < 2; ++hf)
#pragma unroll
    for (int t = 0; t < 4; ++t) acc[hf][t] = zero;

  const int ldRow = tid >> 2;     // 0..63
  const int ldK = (tid & 3) * 8;  // 0,8,16,24

  // issue one async 16B tile chunk per thread
  const __bf16* wsrc = W + (size_t)(nBase + ldRow) * Kd + ldK;
  async_g2l_b128(wsrc + 0, &ldsW[0][ldRow][ldK]);  // prefetch k0 = 0

  const __bf16* arow0 = A + (size_t)(mRow0 + ln) * Kd;
  const __bf16* arow1 = A + (size_t)(mRow0 + 16 + ln) * Kd;
  const int kb = hi * 8;

  for (int k0 = 0; k0 < Kd; k0 += 32) {
    const int buf = (k0 >> 5) & 1;
    wait_async0();      // own prefetch into `buf` landed
    __syncthreads();    // everyone's landed; everyone done reading buf^1
    if (k0 + 32 < Kd)
      async_g2l_b128(wsrc + k0 + 32, &ldsW[buf ^ 1][ldRow][ldK]);

    // A fragments (16x32): lane=row, kb=hi*8, elems 0-7 -> K=kb+e, 8-15 -> 16+kb+e-8
    v16bf a0 = pack16(ldb8(arow0 + k0 + kb), ldb8(arow0 + k0 + kb + 16));
    v16bf a1 = pack16(ldb8(arow1 + k0 + kb), ldb8(arow1 + k0 + kb + 16));

#pragma unroll
    for (int t = 0; t < 4; ++t) {
      // B fragment (32x16): lane=col, elems 0-7 -> K = hi*8 + e, 8-15 -> 16 + hi*8 + e-8
      const __bf16* brow = &ldsW[buf][t * 16 + ln][hi * 8];
      v16bf bfrag = pack16(ldb8(brow), ldb8(brow + 16));
      acc[0][t] = wmma_bf16(a0, bfrag, acc[0][t]);
      acc[1][t] = wmma_bf16(a1, bfrag, acc[1][t]);
    }
  }

#pragma unroll
  for (int t = 0; t < 4; ++t) {
    const int n = nBase + t * 16 + ln;
    const float bb = bias[n];
#pragma unroll
    for (int hf = 0; hf < 2; ++hf) {
#pragma unroll
      for (int r = 0; r < 8; ++r) {
        const int row = mRow0 + hf * 16 + r + hi * 8;  // C layout: M = r + hi*8
        const float v = acc[hf][t][r] + bb;
        if (bf16out)
          outB[(size_t)row * Nd + n] = (__bf16)v;
        else
          outF[(size_t)row * Nd + n] = v;
      }
    }
  }
}

// ---------- RoPE (in place on bf16 Q and K) ----------
__global__ void rope_kernel(__bf16* __restrict__ Q, __bf16* __restrict__ Kb,
                            const int* __restrict__ days) {
  int idx = blockIdx.x * blockDim.x + threadIdx.x;  // B*S*H*32
  int j = idx & 31;
  int h = (idx >> 5) & 15;
  int s = (idx >> 9) & 2047;
  int b = idx >> 20;
  int d = days[b * Sc + s];
  int pos = d < 0 ? -d : d;
  if (pos > 2047) pos = 2047;
  // inv_freq = 10000^(-2j/64) = exp(-(j/32)*ln(10000))
  float freq = (float)pos * __expf(-((float)j * (1.0f / 32.0f)) * 9.210340371976184f);
  float c = __cosf(freq), sn = __sinf(freq);
  size_t base = ((size_t)(b * Sc + s)) * Dc + h * HDc;
  float q1 = (float)Q[base + j], q2 = (float)Q[base + 32 + j];
  Q[base + j] = (__bf16)(q1 * c - q2 * sn);
  Q[base + 32 + j] = (__bf16)(q2 * c + q1 * sn);
  float k1 = (float)Kb[base + j], k2 = (float)Kb[base + 32 + j];
  Kb[base + j] = (__bf16)(k1 * c - k2 * sn);
  Kb[base + 32 + j] = (__bf16)(k2 * c + k1 * sn);
}

// ---------- Flash attention: one wave per (b, h, 16-row q tile) ----------
__global__ __launch_bounds__(256) void flash_attn(
    const __bf16* __restrict__ Q, const __bf16* __restrict__ Kb,
    const __bf16* __restrict__ V, const int* __restrict__ days,
    const float* __restrict__ decay_p, __bf16* __restrict__ O) {
  __shared__ __bf16 ldsV[8][32][72];  // per-wave V tile (32 keys x 64 hd)
  __shared__ __bf16 ldsP[8][16][40];  // per-wave P tile (16 q x 32 keys)

  const int tid = threadIdx.x;
  const int w = tid >> 5, lane = tid & 31, ln = lane & 15, hi = lane >> 4;
  const int wave = blockIdx.x * 8 + w;  // 0..4095
  const int qt = wave & 127;
  const int h = (wave >> 7) & 15;
  const int b = wave >> 11;
  const int qbase = qt * 16;
  const float drate = decay_p[0];

  // Q A-fragments for the two HD halves (K=32 each)
  const __bf16* qrow = Q + ((size_t)(b * Sc + qbase + ln)) * Dc + h * HDc;
  const int kb = hi * 8;
  v16bf aq0 = pack16(ldb8(qrow + kb), ldb8(qrow + kb + 16));
  v16bf aq1 = pack16(ldb8(qrow + 32 + kb), ldb8(qrow + 32 + kb + 16));

  float dq[8];
  int qlim[8];
#pragma unroll
  for (int r = 0; r < 8; ++r) {
    int row = qbase + r + hi * 8;
    qlim[r] = row;
    dq[r] = (float)days[b * Sc + row];
  }
  float m[8], l[8];
  v8f zero = {};
  v8f o[4];
#pragma unroll
  for (int r = 0; r < 8; ++r) { m[r] = -1e30f; l[r] = 0.0f; }
#pragma unroll
  for (int t = 0; t < 4; ++t) o[t] = zero;

  const int nblk = ((qbase + 15) >> 5) + 1;  // causal key-block count
  for (int blk = 0; blk < nblk; ++blk) {
    const int k0 = blk * 32;

    // Prior block's LDS reads must complete before async engine rewrites ldsV.
    wait_ds0();
    // Async-stage V tile (lane = key row, 8x16B per lane); overlaps with scores.
    {
      const __bf16* vrow = V + ((size_t)(b * Sc + k0 + lane)) * Dc + h * HDc;
#pragma unroll
      for (int c = 0; c < 8; ++c)
        async_g2l_b128(vrow + c * 8, &ldsV[w][lane][c * 8]);
    }

    // K^T B-fragments: lane = key col, K dim = hd
    const __bf16* kr0 = Kb + ((size_t)(b * Sc + k0 + ln)) * Dc + h * HDc + hi * 8;
    const __bf16* kr1 = Kb + ((size_t)(b * Sc + k0 + 16 + ln)) * Dc + h * HDc + hi * 8;
    v16bf bk00 = pack16(ldb8(kr0), ldb8(kr0 + 16));
    v16bf bk01 = pack16(ldb8(kr0 + 32), ldb8(kr0 + 48));
    v16bf bk10 = pack16(ldb8(kr1), ldb8(kr1 + 16));
    v16bf bk11 = pack16(ldb8(kr1 + 32), ldb8(kr1 + 48));

    v8f s0 = zero, s1 = zero;
    s0 = wmma_bf16(aq0, bk00, s0);
    s0 = wmma_bf16(aq1, bk01, s0);
    s1 = wmma_bf16(aq0, bk10, s1);
    s1 = wmma_bf16(aq1, bk11, s1);

    const float dk0 = (float)days[b * Sc + k0 + ln];
    const float dk1 = (float)days[b * Sc + k0 + 16 + ln];
    const int key0 = k0 + ln, key1 = k0 + 16 + ln;

    float p0[8], p1[8], scl[8];
#pragma unroll
    for (int r = 0; r < 8; ++r) {
      float v0 = (key0 <= qlim[r])
                     ? s0[r] * 0.125f * __expf(-drate * fabsf(dq[r] - dk0))
                     : -1e30f;
      float v1 = (key1 <= qlim[r])
                     ? s1[r] * 0.125f * __expf(-drate * fabsf(dq[r] - dk1))
                     : -1e30f;
      float mx = fmaxf(v0, v1);
      mx = fmaxf(mx, __shfl_xor(mx, 1, 32));
      mx = fmaxf(mx, __shfl_xor(mx, 2, 32));
      mx = fmaxf(mx, __shfl_xor(mx, 4, 32));
      mx = fmaxf(mx, __shfl_xor(mx, 8, 32));
      const float mnew = fmaxf(m[r], mx);
      const float sc = __expf(m[r] - mnew);
      const float e0 = __expf(v0 - mnew);
      const float e1 = __expf(v1 - mnew);
      float sum = e0 + e1;
      sum += __shfl_xor(sum, 1, 32);
      sum += __shfl_xor(sum, 2, 32);
      sum += __shfl_xor(sum, 4, 32);
      sum += __shfl_xor(sum, 8, 32);
      l[r] = l[r] * sc + sum;
      m[r] = mnew;
      p0[r] = e0; p1[r] = e1; scl[r] = sc;
    }
#pragma unroll
    for (int t = 0; t < 4; ++t) {
#pragma unroll
      for (int r = 0; r < 8; ++r) o[t][r] *= scl[r];
    }
    // P: C-layout -> LDS -> A-layout re-read
#pragma unroll
    for (int r = 0; r < 8; ++r) {
      int M = r + hi * 8;
      ldsP[w][M][ln] = (__bf16)p0[r];
      ldsP[w][M][16 + ln] = (__bf16)p1[r];
    }
    wait_ds0();     // P stores visible
    wait_async0();  // V tile landed
    v16bf pa = pack16(ldb8(&ldsP[w][ln][kb]), ldb8(&ldsP[w][ln][kb + 16]));
#pragma unroll
    for (int t = 0; t < 4; ++t) {
      v16bf bv;
#pragma unroll
      for (int e = 0; e < 16; ++e) bv[e] = ldsV[w][hi * 8 + (e & 7) + (e >> 3) * 16][t * 16 + ln];
      o[t] = wmma_bf16(pa, bv, o[t]);
    }
  }

#pragma unroll
  for (int t = 0; t < 4; ++t) {
#pragma unroll
    for (int r = 0; r < 8; ++r) {
      int row = qbase + r + hi * 8;
      int col = h * HDc + t * 16 + ln;
      O[((size_t)(b * Sc + row)) * Dc + col] = (__bf16)(o[t][r] / l[r]);
    }
  }
}

// ---------- host-side orchestration ----------
extern "C" void kernel_launch(void* const* d_in, const int* in_sizes, int n_in,
                              void* d_out, int out_size, void* d_ws, size_t ws_size,
                              hipStream_t stream) {
  (void)in_sizes; (void)n_in; (void)out_size; (void)ws_size;
  const float* x = (const float*)d_in[0];
  const float* Wq = (const float*)d_in[1];
  const float* bq = (const float*)d_in[2];
  const float* Wk = (const float*)d_in[3];
  const float* bk = (const float*)d_in[4];
  const float* Wv = (const float*)d_in[5];
  const float* bv = (const float*)d_in[6];
  const float* Wo = (const float*)d_in[7];
  const float* bo = (const float*)d_in[8];
  const float* decay = (const float*)d_in[9];
  // d_in[10] = mask (tril) -> implemented as causal test in-kernel
  const int* days = (const int*)d_in[11];
  float* out = (float*)d_out;

  char* ws = (char*)d_ws;
  const size_t NX = (size_t)Bc * Sc * Dc;  // 4 Mi elements
  const size_t NW = (size_t)Dc * Dc;       // 1 Mi elements
  __bf16* xbf = (__bf16*)ws;
  __bf16* wqb = (__bf16*)(ws + NX * 2);
  __bf16* wkb = wqb + NW;
  __bf16* wvb = wkb + NW;
  __bf16* wob = wvb + NW;
  __bf16* qbf = wob + NW;
  __bf16* kbf = qbf + NX;
  __bf16* vbf = kbf + NX;
  __bf16* obf = vbf + NX;

  cast_f32_to_bf16<<<(int)(NX / 256), 256, 0, stream>>>(x, xbf, (int)NX);
  cast_f32_to_bf16<<<(int)(NW / 256), 256, 0, stream>>>(Wq, wqb, (int)NW);
  cast_f32_to_bf16<<<(int)(NW / 256), 256, 0, stream>>>(Wk, wkb, (int)NW);
  cast_f32_to_bf16<<<(int)(NW / 256), 256, 0, stream>>>(Wv, wvb, (int)NW);
  cast_f32_to_bf16<<<(int)(NW / 256), 256, 0, stream>>>(Wo, wob, (int)NW);

  dim3 ggrid(16, 16);  // N/64, M/256
  gemm_nt_bf16<<<ggrid, 256, 0, stream>>>(xbf, wqb, bq, nullptr, qbf, 1);
  gemm_nt_bf16<<<ggrid, 256, 0, stream>>>(xbf, wkb, bk, nullptr, kbf, 1);
  gemm_nt_bf16<<<ggrid, 256, 0, stream>>>(xbf, wvb, bv, nullptr, vbf, 1);

  rope_kernel<<<(Bc * Sc * Hc * 32) / 256, 256, 0, stream>>>(qbf, kbf, days);

  flash_attn<<<512, 256, 0, stream>>>(qbf, kbf, vbf, days, decay, obf);

  gemm_nt_bf16<<<ggrid, 256, 0, stream>>>(obf, wob, bo, out, nullptr, 0);
}
